// NeuralNet_91156385890314
// MI455X (gfx1250) — matrix operand order repaired and checked by reading, not compiled, not run on hardware
//
#include <hip/hip_runtime.h>

// CDNA5 / gfx1250: wave32 WMMA bf16 path.
typedef __attribute__((ext_vector_type(16))) __bf16 v16bf;
typedef __attribute__((ext_vector_type(8)))  float  v8f;

#define BLOCK_THREADS   256
#define WAVES_PER_BLOCK (BLOCK_THREADS / 32)

__device__ __forceinline__ float sigmoidf_fast(float x) {
    return 1.0f / (1.0f + __expf(-x));
}

// One wave handles 16 samples per tile:
//   A (16x96 bf16, 3 chunks of K=32) = [u*m | u | m] rows
//   B (96x16 bf16, 3 chunks)         = w1^T zero-padded from 8 to 16 cols
//   D (16x16 f32)                    = pre-activation h
// Then relu, *w2, cross-lane sum over the 8 valid columns, sigmoid, store.
__global__ __launch_bounds__(BLOCK_THREADS)
void ncf_fwd_wmma_bf16(const int* __restrict__ users,
                       const int* __restrict__ movies,
                       const float* __restrict__ uemb,
                       const float* __restrict__ memb,
                       const float* __restrict__ w1,   // [8, 96] row-major
                       const float* __restrict__ b1,   // [8]
                       const float* __restrict__ w2,   // [8]
                       const float* __restrict__ b2,   // [1]
                       float* __restrict__ out,        // [batch]
                       int batch)
{
    const int lane  = threadIdx.x & 31;
    const int wib   = threadIdx.x >> 5;
    const int wglob = blockIdx.x * WAVES_PER_BLOCK + wib;
    const int nwave = gridDim.x * WAVES_PER_BLOCK;
    const int ntile = (batch + 15) >> 4;

    const int half = lane >> 4;   // 0: lanes 0-15, 1: lanes 16-31
    const int nmod = lane & 15;   // B / D column index for this lane
    const int koff = half * 8;    // B-layout: lane half h holds K 8h..8h+7 in elems 0..7, K 16+8h..16+8h+7 in elems 8..15

    // ---- B fragments (w1 chunks), loaded once per wave, reused over tiles ----
    const bool ncol_valid = (nmod < 8);
    const float* w1r = w1 + (ncol_valid ? nmod : 0) * 96 + koff;  // in-bounds even when padded
    v16bf B0, B1, B2;
#pragma unroll
    for (int j = 0; j < 16; ++j) {
        const int kj = (j < 8) ? j : j + 8;
        B0[j] = ncol_valid ? (__bf16)w1r[kj]      : (__bf16)0.0f;
        B1[j] = ncol_valid ? (__bf16)w1r[32 + kj] : (__bf16)0.0f;
        B2[j] = ncol_valid ? (__bf16)w1r[64 + kj] : (__bf16)0.0f;
    }
    const float b1n = ncol_valid ? b1[nmod] : 0.0f;   // column bias (lane-indexed)
    const float w2n = ncol_valid ? w2[nmod] : 0.0f;   // zero pad kills columns 8..15
    const float b2s = b2[0];

    // A-layout K addressing: lane (row = lane&15) holds K-subsets
    // {kb..kb+7} in elems 0..7 and {kb+16..kb+23} in elems 8..15, kb = 8*half.
    const int kb4 = half * 2;     // float4 index base into the 32-float embedding row

    for (int tile = wglob; tile < ntile; tile += nwave) {
        const int  row   = lane & 15;
        const int  sidx  = tile * 16 + row;
        const bool inb   = (sidx < batch);
        const int  sload = inb ? sidx : 0;

        const int uid = users[sload];
        const int mid = movies[sload];

        const float4* up = (const float4*)(uemb + (size_t)uid * 32);
        const float4* mp = (const float4*)(memb + (size_t)mid * 32);

        // Each lane loads exactly the 64B of u and 64B of m its A-fragment needs
        // (lane L and L+16 cover disjoint halves of sample L's row -> no duplication).
        float4 u0 = up[kb4],     u1 = up[kb4 + 1];
        float4 u2 = up[kb4 + 4], u3 = up[kb4 + 5];
        float4 m0 = mp[kb4],     m1 = mp[kb4 + 1];
        float4 m2 = mp[kb4 + 4], m3 = mp[kb4 + 5];

        float uf[16] = { u0.x, u0.y, u0.z, u0.w,  u1.x, u1.y, u1.z, u1.w,
                         u2.x, u2.y, u2.z, u2.w,  u3.x, u3.y, u3.z, u3.w };
        float mf[16] = { m0.x, m0.y, m0.z, m0.w,  m1.x, m1.y, m1.z, m1.w,
                         m2.x, m2.y, m2.z, m2.w,  m3.x, m3.y, m3.z, m3.w };

        v16bf A0, A1, A2;   // x chunks: [u*m], [u], [m]
#pragma unroll
        for (int j = 0; j < 16; ++j) {
            A0[j] = (__bf16)(uf[j] * mf[j]);
            A1[j] = (__bf16)uf[j];
            A2[j] = (__bf16)mf[j];
        }

        // h = x @ w1^T accumulated over K = 96 in three WMMA steps (f32 acc).
        v8f c = {};
        c = __builtin_amdgcn_wmma_f32_16x16x32_bf16(false, A0, false, B0, (short)0, c, false, false);
        c = __builtin_amdgcn_wmma_f32_16x16x32_bf16(false, A1, false, B1, (short)0, c, false, false);
        c = __builtin_amdgcn_wmma_f32_16x16x32_bf16(false, A2, false, B2, (short)0, c, false, false);

        // D layout: VGPR r = row (r + 8*half'), lane = column. Apply bias+relu+w2
        // then sum the 8 valid columns with xor-shuffles over lane groups of 8.
        float s[8];
#pragma unroll
        for (int r = 0; r < 8; ++r) {
            float h = c[r] + b1n;
            h = fmaxf(h, 0.0f);
            s[r] = h * w2n;
        }
#pragma unroll
        for (int msk = 1; msk <= 4; msk <<= 1) {
#pragma unroll
            for (int r = 0; r < 8; ++r)
                s[r] += __shfl_xor(s[r], msk, 32);
        }

        // Lanes 0..7 hold rows 0..7, lanes 16..23 hold rows 8..15 (divergence only
        // after all WMMAs; reconverges before next iteration).
        if ((lane & 8) == 0) {
            const int r = lane & 7;
            float v = s[0];
#pragma unroll
            for (int j = 1; j < 8; ++j) v = (r == j) ? s[j] : v;
            const int oidx = tile * 16 + half * 8 + r;
            if (oidx < batch) out[oidx] = sigmoidf_fast(v + b2s);
        }
    }
}

extern "C" void kernel_launch(void* const* d_in, const int* in_sizes, int n_in,
                              void* d_out, int out_size, void* d_ws, size_t ws_size,
                              hipStream_t stream) {
    (void)n_in; (void)out_size; (void)d_ws; (void)ws_size;
    const int*   users  = (const int*)  d_in[0];
    const int*   movies = (const int*)  d_in[1];
    const float* uemb   = (const float*)d_in[2];
    const float* memb   = (const float*)d_in[3];
    const float* w1     = (const float*)d_in[4];
    const float* b1     = (const float*)d_in[5];
    const float* w2     = (const float*)d_in[6];
    const float* b2     = (const float*)d_in[7];
    float* out = (float*)d_out;

    const int batch = in_sizes[0];
    const int ntile = (batch + 15) / 16;
    // ~4 tiles per wave to amortize the w1 fragment loads; 8 waves per block.
    int blocks = (ntile + WAVES_PER_BLOCK * 4 - 1) / (WAVES_PER_BLOCK * 4);
    if (blocks < 1) blocks = 1;

    ncf_fwd_wmma_bf16<<<blocks, BLOCK_THREADS, 0, stream>>>(
        users, movies, uemb, memb, w1, b1, w2, b2, out, batch);
}
